// Attention_12472585028102
// MI455X (gfx1250) — matrix operand order repaired and checked by reading, not compiled, not run on hardware
//
#include <hip/hip_runtime.h>

// ---------------------------------------------------------------------------
// SAGAN self-attention for MI455X (gfx1250, wave32, WMMA).
// B=16, CH=64, H=W=64 -> N=4096 queries, M=1024 keys, d=8, dv=32.
// bf16 WMMA (16x16x32) for scores / P*g / W_o, f32 accumulate,
// flash-style two-pass softmax; score tile computed transposed so the
// D-layout of S^T coincides with the A-layout needed by the AV WMMA
// (exp + bf16 convert purely in-lane, no shuffles in the hot loop).
// All padded-K regions are structurally zeroed on exactly one operand,
// so every load is unconditional -> no EXEC churn in the loops.
// ---------------------------------------------------------------------------

typedef __attribute__((ext_vector_type(16))) __bf16 v16bf;
typedef __attribute__((ext_vector_type(8)))  __bf16 v8bf;
typedef __attribute__((ext_vector_type(8)))  float  v8f;

constexpr int Bn = 16;
constexpr int CH = 64;
constexpr int N  = 4096;   // H*W
constexpr int M  = 1024;   // pooled keys

// ---------------------------------------------------------------------------
// Kernel 1: theta = W_theta @ x  per pixel, stored bf16 [B][N][8]
// ---------------------------------------------------------------------------
__global__ __launch_bounds__(256) void proj_theta_kernel(
    const float* __restrict__ x, const float* __restrict__ Wt,
    __bf16* __restrict__ theta)
{
  __shared__ float w[8 * 64];
  for (int i = threadIdx.x; i < 8 * 64; i += 256) w[i] = Wt[i];
  __syncthreads();

  int idx = blockIdx.x * 256 + threadIdx.x;       // 0 .. B*N-1
  int b = idx >> 12;
  int n = idx & (N - 1);
  const float* xs = x + (size_t)b * CH * N + n;

  float acc[8];
#pragma unroll
  for (int o = 0; o < 8; ++o) acc[o] = 0.f;
  for (int c = 0; c < CH; ++c) {
    float xv = xs[(size_t)c * N];
#pragma unroll
    for (int o = 0; o < 8; ++o) acc[o] = fmaf(w[o * 64 + c], xv, acc[o]);
  }
  __bf16* tp = theta + (size_t)idx * 8;
#pragma unroll
  for (int o = 0; o < 8; ++o) tp[o] = (__bf16)acc[o];
}

// ---------------------------------------------------------------------------
// Kernel 2: phi = maxpool2(W_phi@x) -> bf16 [B][M][8]
//           g   = maxpool2(W_g@x)   -> bf16 [B][32][M]   (key-contiguous)
// ---------------------------------------------------------------------------
__global__ __launch_bounds__(256) void proj_pool_kernel(
    const float* __restrict__ x, const float* __restrict__ Wp,
    const float* __restrict__ Wg,
    __bf16* __restrict__ phi, __bf16* __restrict__ g)
{
  __shared__ float w[40 * 64];                       // rows 0..7 phi, 8..39 g
  for (int i = threadIdx.x; i < 8 * 64; i += 256)  w[i] = Wp[i];
  for (int i = threadIdx.x; i < 32 * 64; i += 256) w[512 + i] = Wg[i];
  __syncthreads();

  int idx = blockIdx.x * 256 + threadIdx.x;          // 0 .. B*M-1
  int b  = idx >> 10;
  int m  = idx & (M - 1);
  int h2 = m >> 5;
  int w2 = m & 31;

  float best[40];
#pragma unroll
  for (int o = 0; o < 40; ++o) best[o] = -1e30f;

  for (int dh = 0; dh < 2; ++dh) {
    for (int dw = 0; dw < 2; ++dw) {
      int n = (h2 * 2 + dh) * 64 + (w2 * 2 + dw);
      const float* xs = x + (size_t)b * CH * N + n;
      float acc[40];
#pragma unroll
      for (int o = 0; o < 40; ++o) acc[o] = 0.f;
      for (int c = 0; c < CH; ++c) {
        float xv = xs[(size_t)c * N];
#pragma unroll
        for (int o = 0; o < 40; ++o) acc[o] = fmaf(w[o * 64 + c], xv, acc[o]);
      }
#pragma unroll
      for (int o = 0; o < 40; ++o) best[o] = fmaxf(best[o], acc[o]);
    }
  }

  __bf16* pp = phi + (size_t)idx * 8;
#pragma unroll
  for (int o = 0; o < 8; ++o) pp[o] = (__bf16)best[o];
#pragma unroll
  for (int o = 0; o < 32; ++o)
    g[((size_t)b * 32 + o) * M + m] = (__bf16)best[8 + o];
}

// ---------------------------------------------------------------------------
// Kernel 3: attention + output projection + residual.
// Block = 128 (4 waves); each wave owns one 16-query tile of one batch.
// ---------------------------------------------------------------------------
__global__ __launch_bounds__(128) void attn_kernel(
    const __bf16* __restrict__ theta, const __bf16* __restrict__ phi,
    const __bf16* __restrict__ g, const float* __restrict__ Wo,
    const float* __restrict__ gamma_p, const float* __restrict__ x,
    float* __restrict__ out)
{
  __shared__ float lds_all[4 * 1024];                // 4 KB per wave
  const int lane  = threadIdx.x & 31;
  const int wave  = threadIdx.x >> 5;
  const int b     = blockIdx.y;
  const int qtile = blockIdx.x * 4 + wave;           // 0..255
  const int row16 = lane & 15;
  const int half  = lane >> 4;
  float* lds = lds_all + wave * 1024;

  const v8f zero = {};

  // theta as WMMA B-matrix (32pad x 16q). Only lanes 0-15 elems 0..7 set
  // (K=0..7); all other K rows stay zero, which
  // cancels every padded-K product (see layout note above).
  v16bf bT = {};
  {
    const __bf16* th = theta + ((size_t)b * N + qtile * 16 + row16) * 8;
    v8bf t = *(const v8bf*)th;                       // global_load_b128
#pragma unroll
    for (int e = 0; e < 8; ++e) bT[e] = half ? (__bf16)0.f : t[e];
  }

  // ------------------------- pass 1: max / sum -----------------------------
  float m_run = -1e30f, l_run = 0.f;
  for (int kt = 0; kt < 32; ++kt) {
    const int key0 = kt * 32;
    const __bf16* ph = phi + ((size_t)b * M + key0 + row16) * 8;
    __builtin_prefetch(ph + 32 * 8, 0, 0);           // next key tile
    v8bf p0 = *(const v8bf*)ph;                      // keys key0    ..+15
    v8bf p1 = *(const v8bf*)(ph + 128);              // keys key0+16 ..+31
    v16bf a0 = {}, a1 = {};
#pragma unroll
    for (int e = 0; e < 8; ++e) { a0[e] = p0[e]; a1[e] = p1[e]; }
    v8f s0 = __builtin_amdgcn_wmma_f32_16x16x32_bf16(false, a0, false, bT,
                                                     (short)0, zero, false, false);
    v8f s1 = __builtin_amdgcn_wmma_f32_16x16x32_bf16(false, a1, false, bT,
                                                     (short)0, zero, false, false);
    float tmax = s0[0];
#pragma unroll
    for (int v = 0; v < 8; ++v) {
      tmax = fmaxf(tmax, s0[v]);
      tmax = fmaxf(tmax, s1[v]);
    }
    float m_new = fmaxf(m_run, tmax);
    float ssum = 0.f;
#pragma unroll
    for (int v = 0; v < 8; ++v)
      ssum += __expf(s0[v] - m_new) + __expf(s1[v] - m_new);
    l_run = l_run * __expf(m_run - m_new) + ssum;
    m_run = m_new;
  }
  // merge the (lane, lane^16) pair that shares a query column
  {
    float m_o = __shfl_xor(m_run, 16, 32);
    float l_o = __shfl_xor(l_run, 16, 32);
    float mm  = fmaxf(m_run, m_o);
    float ll  = l_run * __expf(m_run - mm) + l_o * __expf(m_o - mm);
    m_run = mm;
    l_run = 1.0f / ll;                               // inv_l, per-lane scalar
  }

  // ------------------------- pass 2: P @ g ---------------------------------
  v8f acc0 = {}, acc1 = {};
  const __bf16* gb = g + (size_t)b * 32 * M;
  const int c0 = row16, c1 = 16 + row16;
  for (int kt = 0; kt < 32; ++kt) {
    const int key0 = kt * 32;
    const int koff = key0 + half * 16;
    const __bf16* ph = phi + ((size_t)b * M + key0 + row16) * 8;
    __builtin_prefetch(gb + (size_t)c0 * M + koff + 32, 0, 0);
    v8bf p0 = *(const v8bf*)ph;
    v8bf p1 = *(const v8bf*)(ph + 128);
    v16bf a0 = {}, a1 = {};
#pragma unroll
    for (int e = 0; e < 8; ++e) { a0[e] = p0[e]; a1[e] = p1[e]; }
    v8f s0 = __builtin_amdgcn_wmma_f32_16x16x32_bf16(false, a0, false, bT,
                                                     (short)0, zero, false, false);
    v8f s1 = __builtin_amdgcn_wmma_f32_16x16x32_bf16(false, a1, false, bT,
                                                     (short)0, zero, false, false);
    // P in A-layout, normalized in-lane (lane owns query row16)
    v16bf pA;
#pragma unroll
    for (int v = 0; v < 8; ++v) {
      pA[v]     = (__bf16)(__expf(s0[v] - m_run) * l_run);
      pA[8 + v] = (__bf16)(__expf(s1[v] - m_run) * l_run);
    }
    // g tiles as B-matrix: keys 8h..8h+7 and 16+8h..16+8h+7 per lane (16B aligned)
    const __bf16* g0p = gb + (size_t)c0 * M + key0 + half * 8;
    const __bf16* g1p = gb + (size_t)c1 * M + key0 + half * 8;
    v16bf bg0 = __builtin_shufflevector(*(const v8bf*)g0p, *(const v8bf*)(g0p + 16), 0,1,2,3,4,5,6,7,8,9,10,11,12,13,14,15);
    v16bf bg1 = __builtin_shufflevector(*(const v8bf*)g1p, *(const v8bf*)(g1p + 16), 0,1,2,3,4,5,6,7,8,9,10,11,12,13,14,15);
    acc0 = __builtin_amdgcn_wmma_f32_16x16x32_bf16(false, pA, false, bg0,
                                                   (short)0, acc0, false, false);
    acc1 = __builtin_amdgcn_wmma_f32_16x16x32_bf16(false, pA, false, bg1,
                                                   (short)0, acc1, false, false);
  }

  // -------- epilogue: transpose av via LDS, o = W_o @ av, residual ---------
  // av D-layout: lane col c=row16 (+16 for acc1), rows q = half*8+v
#pragma unroll
  for (int v = 0; v < 8; ++v) {
    int q = half * 8 + v;
    lds[q * 32 + row16]      = acc0[v];
    lds[q * 32 + 16 + row16] = acc1[v];
  }
  __syncthreads();

  // av as A-matrix (16q x 32c)
  v16bf avA;
#pragma unroll
  for (int e = 0; e < 8; ++e) {
    avA[e]     = (__bf16)lds[row16 * 32 + half * 8 + e];
    avA[8 + e] = (__bf16)lds[row16 * 32 + 16 + half * 8 + e];
  }
  __syncthreads();

  // 4 output-channel tiles: o[q][ch] = av @ W_o^T ; stash to LDS as [ch][q]
#pragma unroll
  for (int cht = 0; cht < 4; ++cht) {
    int ch = cht * 16 + row16;
    const float* wo = Wo + ch * 32 + half * 8;
    v16bf bw;
#pragma unroll
    for (int e = 0; e < 16; ++e) bw[e] = (__bf16)wo[e < 8 ? e : e + 8];
    v8f ot = __builtin_amdgcn_wmma_f32_16x16x32_bf16(false, avA, false, bw,
                                                     (short)0, zero, false, false);
#pragma unroll
    for (int v = 0; v < 8; ++v) lds[ch * 16 + half * 8 + v] = ot[v];
  }
  __syncthreads();

  // coalesced writeout: each lane owns 2 channel rows of 16 queries (64B)
  float gmm = *gamma_p;
#pragma unroll
  for (int r = 0; r < 2; ++r) {
    int ch = r * 32 + lane;
    const float* xs = x   + ((size_t)b * CH + ch) * N + qtile * 16;
    float*       os = out + ((size_t)b * CH + ch) * N + qtile * 16;
#pragma unroll
    for (int e = 0; e < 16; e += 4) {
      float4 xv = *(const float4*)(xs + e);
      float4 ov;
      ov.x = fmaf(gmm, lds[ch * 16 + e + 0], xv.x);
      ov.y = fmaf(gmm, lds[ch * 16 + e + 1], xv.y);
      ov.z = fmaf(gmm, lds[ch * 16 + e + 2], xv.z);
      ov.w = fmaf(gmm, lds[ch * 16 + e + 3], xv.w);
      *(float4*)(os + e) = ov;
    }
  }
}

// ---------------------------------------------------------------------------
extern "C" void kernel_launch(void* const* d_in, const int* in_sizes, int n_in,
                              void* d_out, int out_size, void* d_ws, size_t ws_size,
                              hipStream_t stream)
{
  (void)in_sizes; (void)n_in; (void)out_size; (void)ws_size;
  const float* x  = (const float*)d_in[0];
  const float* Wt = (const float*)d_in[1];
  const float* Wp = (const float*)d_in[2];
  const float* Wg = (const float*)d_in[3];
  const float* Wo = (const float*)d_in[4];
  const float* gm = (const float*)d_in[5];
  float* out = (float*)d_out;

  char* ws = (char*)d_ws;
  __bf16* theta = (__bf16*)ws;                                   // B*N*8  bf16
  __bf16* phi   = (__bf16*)(ws + (size_t)Bn * N * 8 * 2);        // B*M*8  bf16
  __bf16* g     = (__bf16*)(ws + (size_t)Bn * N * 8 * 2
                               + (size_t)Bn * M * 8 * 2);        // B*32*M bf16

  proj_theta_kernel<<<(Bn * N) / 256, 256, 0, stream>>>(x, Wt, theta);
  proj_pool_kernel<<<(Bn * M) / 256, 256, 0, stream>>>(x, Wp, Wg, phi, g);
  attn_kernel<<<dim3(64, Bn), 128, 0, stream>>>(theta, phi, g, Wo, gm, x, out);
}
